// StockTransformerModel_19473381720728
// MI455X (gfx1250) — matrix operand order repaired and checked by reading, not compiled, not run on hardware
//
#include <hip/hip_runtime.h>

// ---------------------------------------------------------------------------
// MLA transformer forward for MI455X (gfx1250, wave32, WMMA).
// GEMM core: C[M,N](f32) (+)= A[M,K](bf16) x W[N,K](bf16)^T
//  - 128x64 block tile, 8 waves, 32x32 per wave (4 WMMAs / K-step)
//  - double-buffered LDS fed by either:
//      * GLOBAL_LOAD_ASYNC_TO_LDS_B128 (ASYNCcnt, per-thread), or
//      * TENSOR_LOAD_TO_LDS (TDM DMA, TENSORcnt, wave0-issued 2-D tiles)
//  - v_wmma_f32_16x16x32_bf16 matrix pipe, f32 accumulate
// ---------------------------------------------------------------------------

typedef __bf16 bf16_t;
typedef __attribute__((ext_vector_type(16))) __bf16 v16bf;
typedef __attribute__((ext_vector_type(8)))  float  v8f;
typedef __attribute__((ext_vector_type(4)))  unsigned int u32x4;
typedef __attribute__((ext_vector_type(4)))  int i32x4;
typedef __attribute__((ext_vector_type(8)))  int i32x8;

#define T_TOK   1024      // B*S tokens
#define HDIM    768
#define NHEADS  12
#define SEQ     512
#define BATCHB  2
#define NLAYER  12
#define RDIM    512
#define KVAPAD  576       // 544 (R+DR) padded to a multiple of 64
#define NEXP    8
#define IDIM    256
#define SIDIM   1024

// ---------------- small helpers ----------------

__device__ __forceinline__ bf16_t f2bf(float f) {
  union { float f; unsigned u; } a; a.f = f;
  unsigned u = a.u + 0x7FFFu + ((a.u >> 16) & 1u);   // round-to-nearest-even
  union { unsigned short s; bf16_t b; } o; o.s = (unsigned short)(u >> 16);
  return o.b;
}
__device__ __forceinline__ float bf2f(bf16_t b) {
  union { bf16_t b; unsigned short s; } a; a.b = b;
  union { unsigned u; float f; } o; o.u = ((unsigned)a.s) << 16;
  return o.f;
}

__device__ __forceinline__ float warpSum(float v) {
#pragma unroll
  for (int m = 16; m > 0; m >>= 1) v += __shfl_xor(v, m, 32);
  return v;
}
__device__ __forceinline__ float warpMax(float v) {
#pragma unroll
  for (int m = 16; m > 0; m >>= 1) v = fmaxf(v, __shfl_xor(v, m, 32));
  return v;
}
__device__ __forceinline__ float blockSum(float v) {
  __shared__ float s[8]; __shared__ float tot;
  int lane = threadIdx.x & 31, w = threadIdx.x >> 5;
  v = warpSum(v);
  if (lane == 0) s[w] = v;
  __syncthreads();
  if (w == 0) {
    float x = (lane < 8) ? s[lane] : 0.f;
    x = warpSum(x);
    if (lane == 0) tot = x;
  }
  __syncthreads();
  return tot;
}
__device__ __forceinline__ float blockMax(float v) {
  __shared__ float s[8]; __shared__ float tot;
  int lane = threadIdx.x & 31, w = threadIdx.x >> 5;
  v = warpMax(v);
  if (lane == 0) s[w] = v;
  __syncthreads();
  if (w == 0) {
    float x = (lane < 8) ? s[lane] : -3.4e38f;
    x = warpMax(x);
    if (lane == 0) tot = x;
  }
  __syncthreads();
  return tot;
}

// lds byte offset of a generic pointer to __shared__ (low 32 bits of flat addr)
__device__ __forceinline__ unsigned lds_off(const void* p) {
  return (unsigned)(unsigned long long)p;
}

// TDM: DMA a 2-D tile (rows x 32 bf16 elements, row stride ld_elems) from
// global memory into LDS at lds_addr. D# per CDNA5 ISA ch.8 (groups 0/1;
// groups 2/3 zero for a 2-D tensor). Issued once per wave; TENSORcnt-tracked.
__device__ __forceinline__ void tdm_load_2d(unsigned lds_addr, const void* gptr,
                                            int rows, int ld_elems) {
  unsigned long long ga = (unsigned long long)gptr;
  u32x4 g0;
  g0[0] = 1u;                                          // count=1, no gather
  g0[1] = lds_addr;                                    // lds_addr [63:32]
  g0[2] = (unsigned)(ga & 0xFFFFFFFFull);              // global_addr[31:0]
  g0[3] = ((unsigned)((ga >> 32) & 0x01FFFFFFull))     // global_addr[56:32]
        | (2u << 30);                                  // type=2 ("image")
  i32x8 g1;
  g1[0] = (int)(1u << 16);                             // data_size=1 (2 bytes)
  g1[1] = (int)(32u << 16);                            // tensor_dim0 = 32
  g1[2] = (int)(((unsigned)rows & 0xFFFFu) << 16);     // tensor_dim1 = rows
  g1[3] = (int)(32u << 16);                            // tile_dim0 = 32
  g1[4] = (int)((unsigned)rows & 0xFFFFu);             // tile_dim1 = rows
  g1[5] = ld_elems;                                    // tensor_dim0_stride lo32
  g1[6] = 0;                                           // stride hi / dim1_stride lo
  g1[7] = 0;
  i32x4 g2 = {0, 0, 0, 0};
  i32x4 g3 = {0, 0, 0, 0};
#if __clang_major__ >= 23
  i32x8 g4 = {0, 0, 0, 0, 0, 0, 0, 0};
  __builtin_amdgcn_tensor_load_to_lds(g0, g1, g2, g3, g4, 0);
#else
  __builtin_amdgcn_tensor_load_to_lds(g0, g1, g2, g3, 0);
#endif
}

// ---------------- WMMA GEMM ----------------
// Block: 256 threads = 8 waves (4 x 2) -> 128x64 C tile; wave tile 32x32.
// EPI: 0 = store, 1 = accumulate into C, 2 = store + bias[n],
//      3 = C += aux[row*auxStride] * acc  (MoE top-2 weighted combine)
// TDM: 0 = per-thread async-to-LDS staging, 1 = wave0 TDM tensor staging
template<int EPI, int TDM>
__global__ __launch_bounds__(256) void gemm_wmma(
    const bf16_t* __restrict__ A, int lda, long strideA,
    const bf16_t* __restrict__ W, int ldw, long strideW,
    float* __restrict__ C, int ldc, long strideC,
    int M, int N, int K,
    const float* __restrict__ aux, int auxStride)
{
  __shared__ __align__(16) bf16_t As[2][128][32];
  __shared__ __align__(16) bf16_t Bs[2][64][32];

  const int z = blockIdx.z;
  A += (long)z * strideA;
  W += (long)z * strideW;
  C += (long)z * strideC;

  const int m0 = blockIdx.y * 128;
  const int n0 = blockIdx.x * 64;
  const int tid  = threadIdx.x;
  const int lane = tid & 31;
  const int wave = tid >> 5;
  const int wm = wave >> 1;   // 0..3 -> 32-row band
  const int wn = wave & 1;    // 0..1 -> 32-col band

  // async staging addressing: A tile 128x32 (8KB) = 2 x b128 per thread,
  //                           B tile  64x32 (4KB) = 1 x b128 per thread.
  const int arow = tid >> 1;          // 0..127
  const int acol = (tid & 1) * 16;    // 0 or 16
  const int brow = tid >> 2;          // 0..63
  const int bcol = (tid & 3) * 8;     // 0,8,16,24

  // wave-fragment addressing (ISA 16-bit A 16x32 / B 32x16 layouts)
  const int fr   = lane & 15;
  const int aklo = (lane < 16) ? 0 : 8;
  const int bk   = (lane < 16) ? 0 : 8;

  v8f acc[2][2] = {};

  const int nk = K >> 5;   // K/32 steps

  auto issue = [&](int k0, int buf) {
    if (TDM) {
      if (wave == 0) {
        tdm_load_2d(lds_off(&As[buf][0][0]), &A[(long)m0 * lda + k0], 128, lda);
        tdm_load_2d(lds_off(&Bs[buf][0][0]), &W[(long)n0 * ldw + k0], 64, ldw);
      }
    } else {
      const bf16_t* ga = &A[(long)(m0 + arow) * lda + k0 + acol];
      const bf16_t* gb = &W[(long)(n0 + brow) * ldw + k0 + bcol];
      unsigned la = lds_off(&As[buf][arow][acol]);
      unsigned lb = lds_off(&Bs[buf][brow][bcol]);
      asm volatile("global_load_async_to_lds_b128 %0, %1, off"
                   :: "v"(la), "v"(ga) : "memory");
      asm volatile("global_load_async_to_lds_b128 %0, %1, off"
                   :: "v"(la + 16u), "v"(ga + 8) : "memory");
      asm volatile("global_load_async_to_lds_b128 %0, %1, off"
                   :: "v"(lb), "v"(gb) : "memory");
    }
  };

  issue(0, 0);

  for (int it = 0; it < nk; ++it) {
    const int k0  = it << 5;
    const int buf = it & 1;
    const bool hn = (it + 1 < nk);

    if (hn) issue(k0 + 32, buf ^ 1);

    if (k0 + 64 < K) {   // L2 warming two tiles ahead (global_prefetch_b8)
      __builtin_prefetch(&A[(long)(m0 + arow) * lda + k0 + 64 + acol], 0, 1);
      __builtin_prefetch(&W[(long)(n0 + brow) * ldw + k0 + 64 + bcol], 0, 1);
    }

    // wait for tile `it` (leave just-issued next-tile copies in flight)
    if (TDM) {
      if (wave == 0) {
        if (hn) __builtin_amdgcn_s_wait_tensorcnt(2);
        else    __builtin_amdgcn_s_wait_tensorcnt(0);
      }
    } else {
      if (hn) asm volatile("s_wait_asynccnt 0x3" ::: "memory");
      else    asm volatile("s_wait_asynccnt 0x0" ::: "memory");
    }
    __syncthreads();

    union Frag { v16bf v; float4 f[2]; };
    Frag fa[2], fb[2];
#pragma unroll
    for (int i = 0; i < 2; ++i) {
      const int r = wm * 32 + i * 16 + fr;
      fa[i].f[0] = *(const float4*)&As[buf][r][aklo];
      fa[i].f[1] = *(const float4*)&As[buf][r][16 + aklo];
      const int c = wn * 32 + i * 16 + fr;
      fb[i].f[0] = *(const float4*)&Bs[buf][c][bk];
      fb[i].f[1] = *(const float4*)&Bs[buf][c][16 + bk];
    }
#pragma unroll
    for (int i = 0; i < 2; ++i)
#pragma unroll
      for (int j = 0; j < 2; ++j)
        acc[i][j] = __builtin_amdgcn_wmma_f32_16x16x32_bf16(
            false, fa[i].v, false, fb[j].v, (short)0, acc[i][j], false, false);

    __syncthreads();   // all waves done reading `buf` before it is refilled
  }

  // C/D layout: VGPR e -> lanes 0-15: M=e, N=lane; lanes 16-31: M=e+8, N=lane-16
  const int rsel = (lane < 16) ? 0 : 8;
#pragma unroll
  for (int i = 0; i < 2; ++i) {
#pragma unroll
    for (int j = 0; j < 2; ++j) {
      const int rowb = m0 + wm * 32 + i * 16 + rsel;
      const int col  = n0 + wn * 32 + j * 16 + fr;
#pragma unroll
      for (int e = 0; e < 8; ++e) {
        const int row = rowb + e;
        float v = acc[i][j][e];
        float* cp = &C[(long)row * ldc + col];
        if      (EPI == 0) *cp = v;
        else if (EPI == 1) *cp += v;
        else if (EPI == 2) *cp = v + aux[col];
        else if (EPI == 3) *cp += aux[(long)row * auxStride] * v;
      }
    }
  }
}

// ---------------- elementwise kernels ----------------

__global__ void k_f32_to_bf16(const float* __restrict__ in, bf16_t* __restrict__ out, long n) {
  long i = (long)blockIdx.x * blockDim.x + threadIdx.x;
  if (i < n) out[i] = f2bf(in[i]);
}

// rows of `cols` from f32 [ldin] -> rms-normalized bf16 [ldout]
__global__ __launch_bounds__(256) void k_rmsnorm(
    const float* __restrict__ in, int ldin, const float* __restrict__ w,
    bf16_t* __restrict__ out, int ldout, int cols)
{
  const float* x = in + (long)blockIdx.x * ldin;
  float ss = 0.f;
  for (int c = threadIdx.x; c < cols; c += 256) { float v = x[c]; ss += v * v; }
  ss = blockSum(ss);
  float r = rsqrtf(ss / (float)cols + 1e-6f);
  bf16_t* o = out + (long)blockIdx.x * ldout;
  for (int c = threadIdx.x; c < cols; c += 256) o[c] = f2bf(x[c] * r * w[c]);
}

__global__ void k_rope_tables(float* __restrict__ cosT, float* __restrict__ sinT) {
  int s = blockIdx.x, d = threadIdx.x;   // d in 0..31
  float inv = powf(10000.f, -(float)d / 32.f);
  float f = (float)s * inv;
  cosT[s * 32 + d] = cosf(f);
  sinT[s * 32 + d] = sinf(f);
}

// q f32 [t, NH*64] -> qcat bf16 [b,h,s,64] with RoPE applied to dims 32..63
__global__ void k_pack_q(const float* __restrict__ q,
                         const float* __restrict__ cosT, const float* __restrict__ sinT,
                         bf16_t* __restrict__ qcat)
{
  int s = blockIdx.x, h = blockIdx.y, b = blockIdx.z, d = threadIdx.x;
  long t = (long)b * SEQ + s;
  const float* qp = q + t * (NHEADS * 64) + h * 64;
  float v;
  if (d < 32) v = qp[d];
  else {
    int dd = d - 32;
    float x = qp[32 + dd];
    float rot = (dd < 16) ? -qp[32 + dd + 16] : qp[32 + dd - 16];
    v = x * cosT[s * 32 + dd] + rot * sinT[s * 32 + dd];
  }
  qcat[(((long)(b * NHEADS + h)) * SEQ + s) * 64 + d] = f2bf(v);
}

// k_nope from kv f32 [t,1152] (+RoPE'd k_pe from kva f32 [t,KVAPAD]) -> kcat [b,h,s,64]
__global__ void k_pack_k(const float* __restrict__ kv, const float* __restrict__ kva,
                         const float* __restrict__ cosT, const float* __restrict__ sinT,
                         bf16_t* __restrict__ kcat)
{
  int s = blockIdx.x, h = blockIdx.y, b = blockIdx.z, d = threadIdx.x;
  long t = (long)b * SEQ + s;
  float v;
  if (d < 32) v = kv[t * 1152 + h * 96 + d];
  else {
    int dd = d - 32;
    const float* kp = kva + t * KVAPAD + RDIM;
    float x = kp[dd];
    float rot = (dd < 16) ? -kp[dd + 16] : kp[dd - 16];
    v = x * cosT[s * 32 + dd] + rot * sinT[s * 32 + dd];
  }
  kcat[(((long)(b * NHEADS + h)) * SEQ + s) * 64 + d] = f2bf(v);
}

// V^T pack: vt[b,h,d,s] = kv[t, h*96 + 32 + d]  (so PV gemm uses W=[N=64,K=512])
__global__ void k_pack_vt(const float* __restrict__ kv, bf16_t* __restrict__ vt) {
  int s = blockIdx.x, h = blockIdx.y, b = blockIdx.z, d = threadIdx.x;
  long t = (long)b * SEQ + s;
  vt[(((long)(b * NHEADS + h)) * 64 + d) * SEQ + s] = f2bf(kv[t * 1152 + h * 96 + 32 + d]);
}

// causal softmax over scores f32 [z,q,512] -> probs bf16
__global__ __launch_bounds__(256) void k_softmax_causal(
    const float* __restrict__ scores, bf16_t* __restrict__ probs, float scale)
{
  int q = blockIdx.x;
  long base = ((long)blockIdx.y * SEQ + q) * SEQ;
  float loc[2];
  float vmax = -3.4e38f;
#pragma unroll
  for (int i = 0; i < 2; ++i) {
    int k = threadIdx.x + i * 256;
    loc[i] = (k <= q) ? scores[base + k] * scale : -3.4e38f;
    vmax = fmaxf(vmax, loc[i]);
  }
  vmax = blockMax(vmax);
  float sum = 0.f;
#pragma unroll
  for (int i = 0; i < 2; ++i) {
    int k = threadIdx.x + i * 256;
    loc[i] = (k <= q) ? expf(loc[i] - vmax) : 0.f;
    sum += loc[i];
  }
  sum = blockSum(sum);
  float inv = 1.f / sum;
#pragma unroll
  for (int i = 0; i < 2; ++i) {
    int k = threadIdx.x + i * 256;
    probs[base + k] = f2bf(loc[i] * inv);
  }
}

// attnO f32 [z=b*NH+h, s, 64] -> attn bf16 [t, h*64+d]
__global__ void k_pack_attn(const float* __restrict__ attnO, bf16_t* __restrict__ attn) {
  int s = blockIdx.x, h = blockIdx.y, b = blockIdx.z, d = threadIdx.x;
  long t = (long)b * SEQ + s;
  attn[t * (NHEADS * 64) + h * 64 + d] =
      f2bf(attnO[(((long)(b * NHEADS + h)) * SEQ + s) * 64 + d]);
}

__global__ void k_silu_mul(const float* __restrict__ u, const float* __restrict__ v,
                           bf16_t* __restrict__ g, long n) {
  long i = (long)blockIdx.x * blockDim.x + threadIdx.x;
  if (i < n) {
    float a = u[i];
    float s = a / (1.f + expf(-a));
    g[i] = f2bf(s * v[i]);
  }
}

// MoE gate: logits, softmax over 8, top-2 renormalized -> dense wmat [t,8]
__global__ __launch_bounds__(256) void k_gate(const bf16_t* __restrict__ x,
                                              const bf16_t* __restrict__ gw,
                                              float* __restrict__ wmat)
{
  int t = blockIdx.x;
  int w = threadIdx.x >> 5, lane = threadIdx.x & 31;
  const bf16_t* xr = x + (long)t * HDIM;
  const bf16_t* gr = gw + (long)w * HDIM;
  float s = 0.f;
  for (int c = lane; c < HDIM; c += 32) s += bf2f(xr[c]) * bf2f(gr[c]);
  s = warpSum(s);
  __shared__ float logit[NEXP];
  if (lane == 0) logit[w] = s;
  __syncthreads();
  if (threadIdx.x == 0) {
    float m = -3.4e38f;
    for (int e = 0; e < NEXP; ++e) m = fmaxf(m, logit[e]);
    float p[NEXP], sum = 0.f;
    for (int e = 0; e < NEXP; ++e) { p[e] = expf(logit[e] - m); sum += p[e]; }
    for (int e = 0; e < NEXP; ++e) p[e] /= sum;
    int i0 = 0;
    for (int e = 1; e < NEXP; ++e) if (p[e] > p[i0]) i0 = e;
    int i1 = -1;
    for (int e = 0; e < NEXP; ++e) { if (e == i0) continue; if (i1 < 0 || p[e] > p[i1]) i1 = e; }
    float tot = p[i0] + p[i1];
    for (int e = 0; e < NEXP; ++e)
      wmat[t * NEXP + e] = (e == i0) ? p[i0] / tot : ((e == i1) ? p[i1] / tot : 0.f);
  }
}

// final rms on last-token rows + head projection -> out[b]
__global__ __launch_bounds__(256) void k_head(const float* __restrict__ h,
                                              const float* __restrict__ fw,
                                              const float* __restrict__ hw,
                                              const float* __restrict__ hb,
                                              float* __restrict__ out)
{
  int b = blockIdx.x;
  const float* x = h + ((long)b * SEQ + (SEQ - 1)) * HDIM;
  float ss = 0.f;
  for (int c = threadIdx.x; c < HDIM; c += 256) { float v = x[c]; ss += v * v; }
  ss = blockSum(ss);
  float r = rsqrtf(ss / (float)HDIM + 1e-6f);
  float d = 0.f;
  for (int c = threadIdx.x; c < HDIM; c += 256) d += x[c] * r * fw[c] * hw[c];
  d = blockSum(d);
  if (threadIdx.x == 0) out[b] = d + hb[0];
}

// ---------------- host side ----------------

template<int EPI, int TDM>
static void gemmL(hipStream_t st,
                  const void* A, int lda, long sA,
                  const void* W, int ldw, long sW,
                  void* C, int ldc, long sC,
                  int M, int N, int K, int bt,
                  const void* aux, int auxs)
{
  gemm_wmma<EPI, TDM><<<dim3(N / 64, M / 128, bt), dim3(256), 0, st>>>(
      (const bf16_t*)A, lda, sA, (const bf16_t*)W, ldw, sW,
      (float*)C, ldc, sC, M, N, K, (const float*)aux, auxs);
}

static void cvt(hipStream_t st, const float* src, bf16_t* dst, long n) {
  k_f32_to_bf16<<<dim3((unsigned)((n + 255) / 256)), dim3(256), 0, st>>>(src, dst, n);
}

extern "C" void kernel_launch(void* const* d_in, const int* in_sizes, int n_in,
                              void* d_out, int out_size, void* d_ws, size_t ws_size,
                              hipStream_t stream) {
  (void)in_sizes; (void)n_in; (void)out_size;
  const float* in_x  = (const float*)d_in[0];
  const float* in_w  = (const float*)d_in[1];
  const float* in_b  = (const float*)d_in[2];
  const float* anw   = (const float*)d_in[3];
  const float* wq    = (const float*)d_in[4];
  const float* wkva  = (const float*)d_in[5];
  const float* kvnw  = (const float*)d_in[6];
  const float* wkvb  = (const float*)d_in[7];
  const float* wo    = (const float*)d_in[8];
  const float* mnw   = (const float*)d_in[9];
  const float* gw    = (const float*)d_in[10];
  const float* ew1   = (const float*)d_in[11];
  const float* ew2   = (const float*)d_in[12];
  const float* ew3   = (const float*)d_in[13];
  const float* sw1   = (const float*)d_in[14];
  const float* sw2   = (const float*)d_in[15];
  const float* sw3   = (const float*)d_in[16];
  const float* fnw   = (const float*)d_in[17];
  const float* hw    = (const float*)d_in[18];
  const float* hb    = (const float*)d_in[19];

  // ---- workspace carve-out ----
  char* base = (char*)d_ws;
  size_t off = 0;
  auto alloc = [&](size_t bytes) -> void* {
    void* r = base + off;
    off = (off + bytes + 255) & ~(size_t)255;
    return r;
  };
  // per-layer bf16 weight arena (re-converted each layer; ~19MB -> L2 resident)
  bf16_t* wq_b   = (bf16_t*)alloc(2L * HDIM * HDIM);
  bf16_t* wkva_b = (bf16_t*)alloc(2L * KVAPAD * HDIM);   // rows 544..575 unused
  bf16_t* wkvb_b = (bf16_t*)alloc(2L * 1152 * RDIM);
  bf16_t* wo_b   = (bf16_t*)alloc(2L * HDIM * HDIM);
  bf16_t* gw_b   = (bf16_t*)alloc(2L * NEXP * HDIM);
  bf16_t* ew1_b  = (bf16_t*)alloc(2L * NEXP * IDIM * HDIM);
  bf16_t* ew2_b  = (bf16_t*)alloc(2L * NEXP * HDIM * IDIM);
  bf16_t* ew3_b  = (bf16_t*)alloc(2L * NEXP * IDIM * HDIM);
  bf16_t* sw1_b  = (bf16_t*)alloc(2L * SIDIM * HDIM);
  bf16_t* sw2_b  = (bf16_t*)alloc(2L * HDIM * SIDIM);
  bf16_t* sw3_b  = (bf16_t*)alloc(2L * SIDIM * HDIM);
  bf16_t* inw_b  = (bf16_t*)alloc(2L * HDIM * 64);
  bf16_t* x0_b   = (bf16_t*)alloc(2L * T_TOK * 64);
  // activations
  float*  h_f    = (float*)alloc(4L * T_TOK * HDIM);
  bf16_t* x_b    = (bf16_t*)alloc(2L * T_TOK * HDIM);
  float*  q_f    = (float*)alloc(4L * T_TOK * HDIM);
  float*  kva_f  = (float*)alloc(4L * T_TOK * KVAPAD);
  bf16_t* ckv_b  = (bf16_t*)alloc(2L * T_TOK * RDIM);
  float*  kv_f   = (float*)alloc(4L * T_TOK * 1152);
  bf16_t* qcat   = (bf16_t*)alloc(2L * BATCHB * NHEADS * SEQ * 64);
  bf16_t* kcat   = (bf16_t*)alloc(2L * BATCHB * NHEADS * SEQ * 64);
  bf16_t* vt_b   = (bf16_t*)alloc(2L * BATCHB * NHEADS * 64 * SEQ);
  float*  sc_f   = (float*)alloc(4L * BATCHB * NHEADS * SEQ * SEQ);
  bf16_t* pr_b   = (bf16_t*)alloc(2L * BATCHB * NHEADS * SEQ * SEQ);
  float*  ao_f   = (float*)alloc(4L * BATCHB * NHEADS * SEQ * 64);
  bf16_t* at_b   = (bf16_t*)alloc(2L * T_TOK * HDIM);
  bf16_t* xm_b   = (bf16_t*)alloc(2L * T_TOK * HDIM);
  float*  wmat   = (float*)alloc(4L * T_TOK * NEXP);
  float*  u_f    = (float*)alloc(4L * T_TOK * SIDIM);
  float*  v_f    = (float*)alloc(4L * T_TOK * SIDIM);
  bf16_t* g_b    = (bf16_t*)alloc(2L * T_TOK * SIDIM);
  float*  cosT   = (float*)alloc(4L * SEQ * 32);
  float*  sinT   = (float*)alloc(4L * SEQ * 32);
  if (off > ws_size) return;   // workspace too small: bail deterministically

  const int Z = BATCHB * NHEADS;   // 24 attention batches
  const float scale = 0.125f;      // 1/sqrt(DN+DR)=1/8

  // ---- prologue ----
  cvt(stream, in_w, inw_b, (long)HDIM * 64);
  cvt(stream, in_x, x0_b,  (long)T_TOK * 64);
  k_rope_tables<<<dim3(SEQ), dim3(32), 0, stream>>>(cosT, sinT);
  // h = inputs @ in_w^T + in_b   (K=64: shallow pipe, async staging)
  gemmL<2, 0>(stream, x0_b, 64, 0, inw_b, 64, 0, h_f, HDIM, 0,
              T_TOK, HDIM, 64, 1, in_b, 0);

  // ---- layers ----
  for (int l = 0; l < NLAYER; ++l) {
    // convert this layer's weights to bf16
    cvt(stream, wq   + (long)l * HDIM * HDIM,        wq_b,   (long)HDIM * HDIM);
    cvt(stream, wkva + (long)l * 544 * HDIM,         wkva_b, (long)544 * HDIM);
    cvt(stream, wkvb + (long)l * 1152 * RDIM,        wkvb_b, (long)1152 * RDIM);
    cvt(stream, wo   + (long)l * HDIM * HDIM,        wo_b,   (long)HDIM * HDIM);
    cvt(stream, gw   + (long)l * NEXP * HDIM,        gw_b,   (long)NEXP * HDIM);
    cvt(stream, ew1  + (long)l * NEXP * IDIM * HDIM, ew1_b,  (long)NEXP * IDIM * HDIM);
    cvt(stream, ew2  + (long)l * NEXP * HDIM * IDIM, ew2_b,  (long)NEXP * HDIM * IDIM);
    cvt(stream, ew3  + (long)l * NEXP * IDIM * HDIM, ew3_b,  (long)NEXP * IDIM * HDIM);
    cvt(stream, sw1  + (long)l * SIDIM * HDIM,       sw1_b,  (long)SIDIM * HDIM);
    cvt(stream, sw2  + (long)l * HDIM * SIDIM,       sw2_b,  (long)HDIM * SIDIM);
    cvt(stream, sw3  + (long)l * SIDIM * HDIM,       sw3_b,  (long)SIDIM * HDIM);

    // ---- attention ----  (deep-K single-batch GEMMs use the TDM staging path)
    k_rmsnorm<<<dim3(T_TOK), dim3(256), 0, stream>>>(
        h_f, HDIM, anw + (long)l * HDIM, x_b, HDIM, HDIM);
    gemmL<0, 1>(stream, x_b, HDIM, 0, wq_b, HDIM, 0, q_f, HDIM, 0,
                T_TOK, HDIM, HDIM, 1, nullptr, 0);
    gemmL<0, 1>(stream, x_b, HDIM, 0, wkva_b, HDIM, 0, kva_f, KVAPAD, 0,
                T_TOK, KVAPAD, HDIM, 1, nullptr, 0);
    k_rmsnorm<<<dim3(T_TOK), dim3(256), 0, stream>>>(
        kva_f, KVAPAD, kvnw + (long)l * RDIM, ckv_b, RDIM, RDIM);
    gemmL<0, 1>(stream, ckv_b, RDIM, 0, wkvb_b, RDIM, 0, kv_f, 1152, 0,
                T_TOK, 1152, RDIM, 1, nullptr, 0);

    dim3 pg(SEQ, NHEADS, BATCHB);
    k_pack_q <<<pg, dim3(64), 0, stream>>>(q_f, cosT, sinT, qcat);
    k_pack_k <<<pg, dim3(64), 0, stream>>>(kv_f, kva_f, cosT, sinT, kcat);
    k_pack_vt<<<pg, dim3(64), 0, stream>>>(kv_f, vt_b);

    // scores = Qcat @ Kcat^T    (batched over b*h; async staging)
    gemmL<0, 0>(stream, qcat, 64, (long)SEQ * 64, kcat, 64, (long)SEQ * 64,
                sc_f, SEQ, (long)SEQ * SEQ, SEQ, SEQ, 64, Z, nullptr, 0);
    k_softmax_causal<<<dim3(SEQ, Z), dim3(256), 0, stream>>>(sc_f, pr_b, scale);
    // attn = P @ V
    gemmL<0, 0>(stream, pr_b, SEQ, (long)SEQ * SEQ, vt_b, SEQ, (long)64 * SEQ,
                ao_f, 64, (long)SEQ * 64, SEQ, 64, SEQ, Z, nullptr, 0);
    k_pack_attn<<<pg, dim3(64), 0, stream>>>(ao_f, at_b);
    // h += attn @ wo^T
    gemmL<1, 1>(stream, at_b, HDIM, 0, wo_b, HDIM, 0, h_f, HDIM, 0,
                T_TOK, HDIM, HDIM, 1, nullptr, 0);

    // ---- MoE ----
    k_rmsnorm<<<dim3(T_TOK), dim3(256), 0, stream>>>(
        h_f, HDIM, mnw + (long)l * HDIM, xm_b, HDIM, HDIM);
    k_gate<<<dim3(T_TOK), dim3(256), 0, stream>>>(xm_b, gw_b, wmat);

    for (int e = 0; e < NEXP; ++e) {
      gemmL<0, 0>(stream, xm_b, HDIM, 0, ew1_b + (long)e * IDIM * HDIM, HDIM, 0,
                  u_f, IDIM, 0, T_TOK, IDIM, HDIM, 1, nullptr, 0);
      gemmL<0, 0>(stream, xm_b, HDIM, 0, ew3_b + (long)e * IDIM * HDIM, HDIM, 0,
                  v_f, IDIM, 0, T_TOK, IDIM, HDIM, 1, nullptr, 0);
      k_silu_mul<<<dim3((T_TOK * IDIM) / 256), dim3(256), 0, stream>>>(
          u_f, v_f, g_b, (long)T_TOK * IDIM);
      // h += wmat[:,e] * (g @ w2^T)
      gemmL<3, 0>(stream, g_b, IDIM, 0, ew2_b + (long)e * HDIM * IDIM, IDIM, 0,
                  h_f, HDIM, 0, T_TOK, HDIM, IDIM, 1, wmat + e, NEXP);
    }

    // shared expert: h += silu(x@sw1^T)*(x@sw3^T) @ sw2^T  (TDM staging)
    gemmL<0, 1>(stream, xm_b, HDIM, 0, sw1_b, HDIM, 0, u_f, SIDIM, 0,
                T_TOK, SIDIM, HDIM, 1, nullptr, 0);
    gemmL<0, 1>(stream, xm_b, HDIM, 0, sw3_b, HDIM, 0, v_f, SIDIM, 0,
                T_TOK, SIDIM, HDIM, 1, nullptr, 0);
    k_silu_mul<<<dim3((T_TOK * SIDIM) / 256), dim3(256), 0, stream>>>(
        u_f, v_f, g_b, (long)T_TOK * SIDIM);
    gemmL<1, 1>(stream, g_b, SIDIM, 0, sw2_b, SIDIM, 0, h_f, HDIM, 0,
                T_TOK, HDIM, SIDIM, 1, nullptr, 0);
  }

  // ---- head ----
  k_head<<<dim3(BATCHB), dim3(256), 0, stream>>>(h_f, fnw, hw, hb, (float*)d_out);
}
